// SwinTransformerBlock_56195352101037
// MI455X (gfx1250) — compile-verified, matrix operand order checked by reading, not run on hardware
//
#include <hip/hip_runtime.h>
#include <hip/hip_bf16.h>

typedef __bf16 bf16_t;
typedef __attribute__((ext_vector_type(16))) __bf16 v16bf;
typedef __attribute__((ext_vector_type(8)))  float  v8f;

// ---- problem constants (from reference) ----
constexpr int kH = 56, kW = 56, kWS = 7, kShift = 3;
constexpr int kC = 128, kNH = 4, kHD = 32, kN = 49, kNP = 64;
constexpr int kB = 64, kHID = 512;
constexpr int kNWloc = 64;              // windows per image (8x8)
constexpr int kNWIN = kB * kNWloc;      // 4096 total windows
constexpr float kScale = 0.17677669529663687f; // HD^-0.5

// ---- LDS byte-offset map for the attention kernel (dynamic shared, base 0) ----
constexpr unsigned kOffS  = 0;        // scores f32: 4*64*64*4 = 65536
constexpr unsigned kOffA  = 65536;    // A tile bf16: 64 x 136  = 17408
constexpr unsigned kOffQ  = 82944;    // Q bf16: 4 x 64 x 40    = 20480
constexpr unsigned kOffK  = 103424;   // K bf16: 4 x 64 x 40    = 20480
constexpr unsigned kOffVT = 123904;   // V^T bf16: 4 x 32 x 72  = 18432
constexpr unsigned kOffP  = 142336;   // P bf16: 4 x 64 x 72    = 36864
constexpr unsigned kOffO  = 179200;   // O bf16: 64 x 136       = 17408
constexpr unsigned kSmemA = 196608;   // total 192 KB (< 320 KB/WGP)

// ---- LDS byte-offset map for the MLP kernel ----
constexpr unsigned kOffR  = 0;        // residual f32: 64 x 132 = 33792
constexpr unsigned kOffMu = 33792;
constexpr unsigned kOffRs = 34048;
constexpr unsigned kOffX  = 34304;    // LN2 bf16: 64 x 136     = 17408
constexpr unsigned kOffH  = 51712;    // hidden bf16: 64 x 520  = 66560
constexpr unsigned kSmemM = 118272;   // 115.5 KB

__device__ __forceinline__ v8f wmma_bf16(v16bf a, v16bf b, v8f c) {
  // D = A(16x32) * B(32x16) + C, fp32 accumulate -> v_wmma_f32_16x16x32_bf16
  return __builtin_amdgcn_wmma_f32_16x16x32_bf16(false, a, false, b, (short)0, c,
                                                 false, false);
}

// CDNA5 async global->LDS copy (16 B per lane), tracked by ASYNCcnt.
__device__ __forceinline__ void async_g2l_b128(unsigned lds_byte_off,
                                               const void* gsrc) {
  asm volatile("global_load_async_to_lds_b128 %0, %1, off"
               :
               : "v"(lds_byte_off), "v"((unsigned long long)(uintptr_t)gsrc)
               : "memory");
}
__device__ __forceinline__ void wait_asynccnt0() {
  asm volatile("s_wait_asynccnt 0x0" ::: "memory");
}

// Load a 16x32 bf16 fragment (A: rows = M, or B: rows = N of a row-major
// [N][K] weight).  ISA layout: lane&15 -> row, lane>>4 -> K half; each lane
// reads K = kb..kb+7 and kb+16..kb+23 (two contiguous 16 B chunks).
__device__ __forceinline__ v16bf frag_ld(const bf16_t* base, int row0, int stride,
                                         int k0) {
  const int lane = threadIdx.x & 31;
  const bf16_t* p = base + (size_t)(row0 + (lane & 15)) * stride + k0 +
                    ((lane >> 4) << 3);
  v16bf f;
#pragma unroll
  for (int j = 0; j < 8; ++j) {
    f[j]     = p[j];
    f[j + 8] = p[j + 16];
  }
  return f;
}

__device__ __forceinline__ v8f vzero8() {
  v8f z = {0.f, 0.f, 0.f, 0.f, 0.f, 0.f, 0.f, 0.f};
  return z;
}

// ---------------------------------------------------------------------------
// Kernel 0: fp32 -> bf16 weight conversion
// ---------------------------------------------------------------------------
__global__ void cvt_bf16_kernel(const float* __restrict__ src,
                                bf16_t* __restrict__ dst, int n) {
  int i = blockIdx.x * 256 + threadIdx.x;
  if (i < n) dst[i] = (bf16_t)src[i];
}

// ---------------------------------------------------------------------------
// Kernel 1: LN1 + roll(-SHIFT) + window partition -> xw bf16 [4096][64][128]
// rows 49..63 zero-padded.  One block per window, one wave per token.
// ---------------------------------------------------------------------------
__global__ void prep_windows_kernel(const float* __restrict__ x,
                                    const float* __restrict__ n1w,
                                    const float* __restrict__ n1b,
                                    bf16_t* __restrict__ xw) {
  const int w = blockIdx.x;            // 0..4095
  const int b = w >> 6;
  const int wh = (w >> 3) & 7;
  const int ww = w & 7;
  const int wv = __builtin_amdgcn_readfirstlane(threadIdx.x >> 5);
  const int lane = threadIdx.x & 31;

  // zero the pad rows
  for (int s = 49 + wv; s < kNP; s += 8)
    for (int c = lane; c < kC; c += 32)
      xw[((size_t)w * kNP + s) * kC + c] = (bf16_t)0.0f;

  for (int s = wv; s < kN; s += 8) {
    const int i = s / 7, j = s % 7;
    const int hs = (wh * 7 + i + kShift) % kH;
    const int ws = (ww * 7 + j + kShift) % kW;
    const float* xp = x + ((size_t)b * (kH * kW) + hs * kW + ws) * kC;
    float4 v = ((const float4*)xp)[lane];
    float s1 = v.x + v.y + v.z + v.w;
    float s2 = v.x * v.x + v.y * v.y + v.z * v.z + v.w * v.w;
#pragma unroll
    for (int o = 16; o >= 1; o >>= 1) {
      s1 += __shfl_xor(s1, o, 32);
      s2 += __shfl_xor(s2, o, 32);
    }
    const float mu = s1 * (1.0f / kC);
    const float rs = rsqrtf(s2 * (1.0f / kC) - mu * mu + 1e-5f);
    bf16_t* op = xw + ((size_t)w * kNP + s) * kC + lane * 4;
    const int c0 = lane * 4;
    op[0] = (bf16_t)((v.x - mu) * rs * n1w[c0 + 0] + n1b[c0 + 0]);
    op[1] = (bf16_t)((v.y - mu) * rs * n1w[c0 + 1] + n1b[c0 + 1]);
    op[2] = (bf16_t)((v.z - mu) * rs * n1w[c0 + 2] + n1b[c0 + 2]);
    op[3] = (bf16_t)((v.w - mu) * rs * n1w[c0 + 3] + n1b[c0 + 3]);
  }
}

// ---------------------------------------------------------------------------
// Kernel 2: full window attention (qkv -> QK^T -> softmax -> PV -> proj).
// A tile staged ONCE via GLOBAL_LOAD_ASYNC_TO_LDS_B128; the alpha channel
// masks shrink monotonically (128->64->32) so masking is incremental zeroing.
// One block (8 waves) per window.  All GEMMs via v_wmma_f32_16x16x32_bf16.
// Wave id pinned to an SGPR so job decomposition / Q|K|V routing is scalar.
// ---------------------------------------------------------------------------
__global__ void window_attn_kernel(const bf16_t* __restrict__ xw,
                                   const bf16_t* __restrict__ qkvW,   // [384][128]
                                   const float* __restrict__ qkvB,
                                   const bf16_t* __restrict__ projW,  // [128][128]
                                   const float* __restrict__ projB,
                                   const float* __restrict__ rbt,     // [169][4]
                                   const int* __restrict__ relIdx,    // [49*49]
                                   const float* __restrict__ amask,   // [64][49][49]
                                   const float* __restrict__ gw,      // [4]
                                   float* __restrict__ ybuf) {        // [4096][49][128]
  extern __shared__ char smem[];
  float*  sS  = (float*)(smem + kOffS);
  bf16_t* sA  = (bf16_t*)(smem + kOffA);
  bf16_t* sQ  = (bf16_t*)(smem + kOffQ);
  bf16_t* sK  = (bf16_t*)(smem + kOffK);
  bf16_t* sVT = (bf16_t*)(smem + kOffVT);
  bf16_t* sP  = (bf16_t*)(smem + kOffP);
  bf16_t* sO  = (bf16_t*)(smem + kOffO);

  const int w = blockIdx.x;
  const int wloc = w & 63;
  const bool small = (w >= kNWIN / 2);
  const int tid = threadIdx.x;
  const int wvs = __builtin_amdgcn_readfirstlane(tid >> 5);  // SGPR wave id
  const int lane = tid & 31;
  const int mb = (lane >> 4) << 3;   // accumulator row offset (0 or 8)
  const int nc = lane & 15;          // accumulator column

  // ---- async-stage full A tile: 64 rows x 256 B -> LDS stride 272 B ----
  {
    const char* gsrc = (const char*)(xw + (size_t)w * kNP * kC);
#pragma unroll
    for (int it = 0; it < 4; ++it) {
      const int e = tid + it * 256;          // 16-byte chunk id, 0..1023
      const int r = e >> 4, cc = e & 15;     // row, 16B chunk within row
      async_g2l_b128(kOffA + (unsigned)(r * 272 + cc * 16),
                     gsrc + r * 256 + cc * 16);
    }
    wait_asynccnt0();
  }
  __syncthreads();

  v8f accP[4] = {vzero8(), vzero8(), vzero8(), vzero8()};

  const int nA = small ? 3 : 1;      // alpha = 1,2,4 (alpha=8 > NH skipped)
  for (int ai = 0; ai < nA; ++ai) {
    const int climit = small ? (kC >> ai) : kC;
    const float g = small ? gw[ai] : 1.0f;

    // incremental channel masking: zero cols [climit, 2*climit); climit = 1<<sh
    if (ai > 0) {
      const int sh = 7 - ai;
      for (int idx = tid; idx < (kNP << sh); idx += 256) {
        const int r = idx >> sh, c = climit + (idx & (climit - 1));
        sA[r * 136 + c] = (bf16_t)0.0f;
      }
      __syncthreads();
    }

    // ---- QKV GEMM: M=64, N=384, K=128 (96 tile jobs, 12 per wave) ----
    for (int jj = 0; jj < 12; ++jj) {
      const int job = wvs * 12 + jj;          // scalar
      const int mt = job / 24, nt = job % 24; // scalar
      v8f acc = vzero8();
#pragma unroll
      for (int kt = 0; kt < 4; ++kt) {
        v16bf a = frag_ld(sA, mt * 16, 136, kt * 32);
        v16bf bq = frag_ld(qkvW, nt * 16, 128, kt * 32);
        acc = wmma_bf16(a, bq, acc);
      }
      const int ngi = nt * 16 + nc;
      const float bias = qkvB[ngi];
      // scalar branch: whole 16-col tile lies inside Q, K or V
      if (nt < 8) {                      // Q (scaled)
        bf16_t* dst = sQ + (ngi >> 5) * (64 * 40) + (ngi & 31);
#pragma unroll
        for (int i = 0; i < 8; ++i)
          dst[(mt * 16 + mb + i) * 40] = (bf16_t)((acc[i] + bias) * kScale);
      } else if (nt < 16) {              // K
        const int hch = ngi - 128;
        bf16_t* dst = sK + (hch >> 5) * (64 * 40) + (hch & 31);
#pragma unroll
        for (int i = 0; i < 8; ++i)
          dst[(mt * 16 + mb + i) * 40] = (bf16_t)(acc[i] + bias);
      } else {                           // V stored transposed: [hd][key]
        const int hch = ngi - 256;
        bf16_t* dst = sVT + (hch >> 5) * (32 * 72) + (hch & 31) * 72;
#pragma unroll
        for (int i = 0; i < 8; ++i)
          dst[mt * 16 + mb + i] = (bf16_t)(acc[i] + bias);
      }
    }
    __syncthreads();

    // ---- scores: per head Q(64x32) @ K^T -> 64x64 (64 jobs, 8/wave) ----
    for (int jj = 0; jj < 8; ++jj) {
      const int job = wvs * 8 + jj;
      const int h = job >> 4, mt = (job >> 2) & 3, nt = job & 3;
      v16bf a = frag_ld(sQ + h * (64 * 40), mt * 16, 40, 0);
      v16bf bq = frag_ld(sK + h * (64 * 40), nt * 16, 40, 0);
      v8f acc = wmma_bf16(a, bq, vzero8());
#pragma unroll
      for (int i = 0; i < 8; ++i)
        sS[h * 4096 + (mt * 16 + mb + i) * 64 + nt * 16 + nc] = acc[i];
    }
    __syncthreads();

    // ---- softmax with rel-pos bias + shift mask (one row per thread) ----
    {
      const int h = tid >> 6, m = tid & 63;
      if (m < kN) {
        float mx = -3e38f;
        for (int n = 0; n < 64; ++n) {
          float s;
          if (n < kN) {
            s = sS[h * 4096 + m * 64 + n];
            const int ri = relIdx[m * kN + n];
            s += rbt[ri * kNH + h];
            s += amask[(size_t)wloc * kN * kN + m * kN + n];
          } else {
            s = -1e30f;
          }
          sS[h * 4096 + m * 64 + n] = s;
          mx = fmaxf(mx, s);
        }
        float sum = 0.0f;
        for (int n = 0; n < 64; ++n) sum += __expf(sS[h * 4096 + m * 64 + n] - mx);
        const float inv = 1.0f / sum;
        for (int n = 0; n < 64; ++n)
          sP[h * (64 * 72) + m * 72 + n] =
              (bf16_t)(__expf(sS[h * 4096 + m * 64 + n] - mx) * inv);
      } else {
        for (int n = 0; n < 64; ++n)
          sP[h * (64 * 72) + m * 72 + n] = (bf16_t)0.0f;
      }
    }
    __syncthreads();

    // ---- O = P(64x64) @ V(64x32) per head (32 jobs, 4/wave) ----
    for (int jj = 0; jj < 4; ++jj) {
      const int job = wvs * 4 + jj;
      const int h = job >> 3, mt = (job >> 1) & 3, nt = job & 1;
      v8f acc = vzero8();
#pragma unroll
      for (int kt = 0; kt < 2; ++kt) {
        v16bf a = frag_ld(sP + h * (64 * 72), mt * 16, 72, kt * 32);
        v16bf bq = frag_ld(sVT + h * (32 * 72), nt * 16, 72, kt * 32);
        acc = wmma_bf16(a, bq, acc);
      }
#pragma unroll
      for (int i = 0; i < 8; ++i)
        sO[(mt * 16 + mb + i) * 136 + h * 32 + nt * 16 + nc] = (bf16_t)acc[i];
    }
    __syncthreads();

    // ---- proj: M=64, N=128, K=128; accumulate g * mask * (o@W^T + b) ----
    for (int jl = 0; jl < 4; ++jl) {
      const int job = wvs * 4 + jl;
      const int mt = job >> 3, nt = job & 7;
      v8f acc = vzero8();
#pragma unroll
      for (int kt = 0; kt < 4; ++kt) {
        v16bf a = frag_ld(sO, mt * 16, 136, kt * 32);
        v16bf bq = frag_ld(projW, nt * 16, 128, kt * 32);
        acc = wmma_bf16(a, bq, acc);
      }
      const int col = nt * 16 + nc;
      const float pb = projB[col];
      const float msk = (col < climit) ? g : 0.0f;
#pragma unroll
      for (int i = 0; i < 8; ++i) accP[jl][i] += (acc[i] + pb) * msk;
    }
    __syncthreads();
  }

  // ---- write window attention output (valid rows only) ----
  for (int jl = 0; jl < 4; ++jl) {
    const int job = wvs * 4 + jl;
    const int mt = job >> 3, nt = job & 7;
    const int col = nt * 16 + nc;
#pragma unroll
    for (int i = 0; i < 8; ++i) {
      const int row = mt * 16 + mb + i;
      if (row < kN)
        ybuf[((size_t)w * kN + row) * kC + col] = accP[jl][i];
    }
  }
}

// ---------------------------------------------------------------------------
// Kernel 3: residual gather (reverse roll) + LN2 + fused MLP (fc1-GELU-fc2,
// hidden kept in LDS).  LN2 tile staged once; alpha masks applied by
// incremental zeroing.  One block per 64 tokens.
// ---------------------------------------------------------------------------
__global__ void mlp_kernel(const float* __restrict__ x,
                           const float* __restrict__ ybuf,
                           const float* __restrict__ n2w,
                           const float* __restrict__ n2b,
                           const bf16_t* __restrict__ f1W,  // [512][128]
                           const float* __restrict__ f1B,
                           const bf16_t* __restrict__ f2W,  // [128][512]
                           const float* __restrict__ f2B,
                           const float* __restrict__ gw,
                           float* __restrict__ out) {
  extern __shared__ char smem[];
  float*  sR  = (float*)(smem + kOffR);
  float*  sMu = (float*)(smem + kOffMu);
  float*  sRs = (float*)(smem + kOffRs);
  bf16_t* sX  = (bf16_t*)(smem + kOffX);
  bf16_t* sH  = (bf16_t*)(smem + kOffH);

  const int blk = blockIdx.x;
  const int b = blk / 49;
  const int t0 = (blk % 49) * 64;
  const bool small = (b >= kB / 2);
  const int tid = threadIdx.x;
  const int wvs = __builtin_amdgcn_readfirstlane(tid >> 5);  // SGPR wave id
  const int lane = tid & 31;
  const int mb = (lane >> 4) << 3;
  const int nc = lane & 15;

  // ---- stage residual: x + roll(+SHIFT)(window-reverse(attn)) ----
  for (int idx = tid; idx < kNP * kC; idx += 256) {
    const int r = idx >> 7, c = idx & 127;
    const int p = t0 + r;
    const int hh = p / kW, wc = p % kW;
    int hs = hh - kShift; if (hs < 0) hs += kH;
    int ws = wc - kShift; if (ws < 0) ws += kW;
    const int win = (b << 6) | ((hs / 7) << 3) | (ws / 7);
    const int slot = (hs % 7) * 7 + (ws % 7);
    sR[r * 132 + c] = x[((size_t)b * (kH * kW) + p) * kC + c] +
                      ybuf[((size_t)win * kN + slot) * kC + c];
  }
  __syncthreads();

  // ---- LN2 stats: 4 lanes per token, quad reduce ----
  {
    const int tok = tid >> 2, q = tid & 3;
    float s1 = 0.f, s2 = 0.f;
    for (int c = q * 32; c < q * 32 + 32; ++c) {
      const float v = sR[tok * 132 + c];
      s1 += v; s2 += v * v;
    }
    s1 += __shfl_xor(s1, 1, 32); s2 += __shfl_xor(s2, 1, 32);
    s1 += __shfl_xor(s1, 2, 32); s2 += __shfl_xor(s2, 2, 32);
    if (q == 0) {
      const float mu = s1 * (1.0f / kC);
      sMu[tok] = mu;
      sRs[tok] = rsqrtf(s2 * (1.0f / kC) - mu * mu + 1e-5f);
    }
  }
  __syncthreads();

  // ---- stage LN2 output once (unmasked) ----
  for (int idx = tid; idx < kNP * kC; idx += 256) {
    const int r = idx >> 7, c = idx & 127;
    sX[r * 136 + c] =
        (bf16_t)((sR[r * 132 + c] - sMu[r]) * sRs[r] * n2w[c] + n2b[c]);
  }
  __syncthreads();

  v8f accM[4] = {vzero8(), vzero8(), vzero8(), vzero8()};

  const int nA = small ? 3 : 1;
  for (int ai = 0; ai < nA; ++ai) {
    const int climit = small ? (kC >> ai) : kC;
    const float g = small ? gw[ai] : 1.0f;

    // incremental channel masking of LN2 tile; climit = 1<<sh
    if (ai > 0) {
      const int sh = 7 - ai;
      for (int idx = tid; idx < (kNP << sh); idx += 256) {
        const int r = idx >> sh, c = climit + (idx & (climit - 1));
        sX[r * 136 + c] = (bf16_t)0.0f;
      }
      __syncthreads();
    }

    // fc1 + exact GELU: M=64, N=512, K=128 (128 jobs, 16/wave)
    for (int jj = 0; jj < 16; ++jj) {
      const int job = wvs * 16 + jj;
      const int mt = job >> 5, nt = job & 31;
      v8f acc = vzero8();
#pragma unroll
      for (int kt = 0; kt < 4; ++kt) {
        v16bf a = frag_ld(sX, mt * 16, 136, kt * 32);
        v16bf bq = frag_ld(f1W, nt * 16, 128, kt * 32);
        acc = wmma_bf16(a, bq, acc);
      }
      const int col = nt * 16 + nc;
      const float bb = f1B[col];
#pragma unroll
      for (int i = 0; i < 8; ++i) {
        float v = acc[i] + bb;
        v = 0.5f * v * (1.0f + erff(v * 0.70710678118654752f));
        sH[(mt * 16 + mb + i) * 520 + col] = (bf16_t)v;
      }
    }
    __syncthreads();

    // fc2: M=64, N=128, K=512 (32 jobs, 4/wave), alpha-accumulate
    for (int jl = 0; jl < 4; ++jl) {
      const int job = wvs * 4 + jl;
      const int mt = job >> 3, nt = job & 7;
      v8f acc = vzero8();
#pragma unroll
      for (int kt = 0; kt < 16; ++kt) {
        v16bf a = frag_ld(sH, mt * 16, 520, kt * 32);
        v16bf bq = frag_ld(f2W, nt * 16, 512, kt * 32);
        acc = wmma_bf16(a, bq, acc);
      }
      const int col = nt * 16 + nc;
      const float bb = f2B[col];
      const float msk = (col < climit) ? g : 0.0f;
#pragma unroll
      for (int i = 0; i < 8; ++i) accM[jl][i] += (acc[i] + bb) * msk;
    }
    __syncthreads();
  }

  // ---- final: out = residual + mlp ----
  for (int jl = 0; jl < 4; ++jl) {
    const int job = wvs * 4 + jl;
    const int mt = job >> 3, nt = job & 7;
    const int col = nt * 16 + nc;
#pragma unroll
    for (int i = 0; i < 8; ++i) {
      const int row = mt * 16 + mb + i;
      const int p = t0 + row;
      out[((size_t)b * (kH * kW) + p) * kC + col] = sR[row * 132 + col] + accM[jl][i];
    }
  }
}

// ---------------------------------------------------------------------------
extern "C" void kernel_launch(void* const* d_in, const int* in_sizes, int n_in,
                              void* d_out, int out_size, void* d_ws, size_t ws_size,
                              hipStream_t stream) {
  (void)in_sizes; (void)n_in; (void)out_size; (void)ws_size;
  const float* x     = (const float*)d_in[0];
  const float* gw    = (const float*)d_in[1];
  const float* n1w   = (const float*)d_in[2];
  const float* n1b   = (const float*)d_in[3];
  const float* qkvw  = (const float*)d_in[4];
  const float* qkvb  = (const float*)d_in[5];
  const float* rbt   = (const float*)d_in[6];
  const float* projw = (const float*)d_in[7];
  const float* projb = (const float*)d_in[8];
  const float* amask = (const float*)d_in[9];
  const float* n2w   = (const float*)d_in[10];
  const float* n2b   = (const float*)d_in[11];
  const float* f1w   = (const float*)d_in[12];
  const float* f1b   = (const float*)d_in[13];
  const float* f2w   = (const float*)d_in[14];
  const float* f2b   = (const float*)d_in[15];
  const int*   relidx = (const int*)d_in[16];
  float* out = (float*)d_out;

  // workspace carve
  char* ws = (char*)d_ws;
  size_t off = 0;
  bf16_t* xw = (bf16_t*)(ws + off);      off += (size_t)kNWIN * kNP * kC * 2;   // 64 MB
  float* ybuf = (float*)(ws + off);      off += (size_t)kNWIN * kN * kC * 4;    // 98 MB
  bf16_t* qkvWb = (bf16_t*)(ws + off);   off += (size_t)3 * kC * kC * 2;
  bf16_t* projWb = (bf16_t*)(ws + off);  off += (size_t)kC * kC * 2;
  bf16_t* f1Wb = (bf16_t*)(ws + off);    off += (size_t)kHID * kC * 2;
  bf16_t* f2Wb = (bf16_t*)(ws + off);    off += (size_t)kC * kHID * 2;

  // 0) weight conversion to bf16
  cvt_bf16_kernel<<<(3 * kC * kC + 255) / 256, 256, 0, stream>>>(qkvw, qkvWb, 3 * kC * kC);
  cvt_bf16_kernel<<<(kC * kC + 255) / 256, 256, 0, stream>>>(projw, projWb, kC * kC);
  cvt_bf16_kernel<<<(kHID * kC + 255) / 256, 256, 0, stream>>>(f1w, f1Wb, kHID * kC);
  cvt_bf16_kernel<<<(kC * kHID + 255) / 256, 256, 0, stream>>>(f2w, f2Wb, kC * kHID);

  // 1) LN1 + shift + partition
  prep_windows_kernel<<<kNWIN, 256, 0, stream>>>(x, n1w, n1b, xw);

  // 2) window attention (192 KB dynamic LDS, fits in 320 KB/WGP)
  window_attn_kernel<<<kNWIN, 256, kSmemA, stream>>>(
      xw, qkvWb, qkvb, projWb, projb, rbt, relidx, amask, gw, ybuf);

  // 3) fused residual + LN2 + MLP (115.5 KB dynamic LDS)
  mlp_kernel<<<kB * 49, 256, kSmemM, stream>>>(
      x, ybuf, n2w, n2b, f1Wb, f1b, f2Wb, f2b, gw, out);
}
